// LSTMNet_38147899523196
// MI455X (gfx1250) — matrix operand order repaired and checked by reading, not compiled, not run on hardware
//
#include <hip/hip_runtime.h>

// ---------------------------------------------------------------------------
// Problem dims (fixed by the reference)
// ---------------------------------------------------------------------------
#define BB    16          // batch
#define SS    16          // sentences per side
#define LL    64          // timesteps
#define EE    512         // embedding dim (= K of input GEMM)
#define HH    512         // hidden dim   (= K of recurrent GEMM)
#define NN    512         // packed batch = B*2*S
#define G4H   2048        // 4*H
#define O1    256

typedef _Float16 half_t;
typedef __attribute__((ext_vector_type(16))) _Float16 v16h;
typedef __attribute__((ext_vector_type(8)))  float    v8f;
typedef __attribute__((ext_vector_type(4)))  int      v4i;

#define LPAD 36           // LDS row pitch in halves (32 + 4 pad)
#define GPAD 132          // LDS gate row pitch in floats (128 + 4 pad)
#define NCH  (EE / 32)    // K chunks

#define AS_GLOBAL __attribute__((address_space(1)))
#define AS_LDS    __attribute__((address_space(3)))

// ---- CDNA5 data-movement feature detection --------------------------------
#if __has_builtin(__builtin_amdgcn_tensor_load_to_lds)
  #define USE_TDM 1
#else
  #define USE_TDM 0
#endif
#if __has_builtin(__builtin_amdgcn_global_load_async_to_lds_b128)
  #define USE_ASYNC 1
#else
  #define USE_ASYNC 0
#endif

__device__ __forceinline__ float sigf(float x) { return 1.0f / (1.0f + __expf(-x)); }

__device__ __forceinline__ unsigned lds_offset(const void* p) {
    return (unsigned)(size_t)(AS_LDS const char*)p;
}

#if USE_TDM
// One 2D TDM descriptor: load a [128 rows x 32 halves] tile whose rows are
// 512 halves apart in global memory, into LDS with a hardware pad of
// 2 DWORDs after every 16 DWORDs (row pitch 72B == LPAD halves).
__device__ __forceinline__ void tdm_load_tile(const half_t* gsrc, unsigned ldsOff,
                                              int tensor_rows) {
    typedef __attribute__((ext_vector_type(4))) unsigned int u32x4;
    typedef __attribute__((ext_vector_type(8))) int          i32x8;
    unsigned long long ga = (unsigned long long)(size_t)gsrc;
    u32x4 g0;
    g0[0] = 1u;                                            // count=1, user mode
    g0[1] = ldsOff;                                        // lds_addr
    g0[2] = (unsigned)(ga & 0xFFFFFFFFu);                  // global_addr[31:0]
    g0[3] = (unsigned)((ga >> 32) & 0x01FFFFFFu)           // global_addr[56:32]
          | (2u << 30);                                    // type = 2 (image)
    i32x8 g1;
    g1[0] = (1 << 16)      // data_size = 1 -> 2 bytes
          | (1 << 20)      // pad_enable
          | (3 << 22)      // pad_interval: 16 DWORDs (one 64B row)
          | (1 << 25);     // pad_amount: 2 DWORDs (8B) -> pitch 72B
    g1[1] = (EE & 0xFFFF) << 16;           // tensor_dim0 = 512 (bits 79:48 lo)
    g1[2] = (tensor_rows & 0xFFFF) << 16;  // tensor_dim1 lo16 (bits 111:80)
    g1[3] = (32 << 16);                    // tile_dim0 = 32 halves
    g1[4] = 128;                           // tile_dim1 = 128 rows
    g1[5] = EE;                            // tensor_dim0_stride = 512 halves
    g1[6] = 0;
    g1[7] = 0;
    v4i z = {0, 0, 0, 0};
#if defined(__clang_major__) && (__clang_major__ >= 23)
    i32x8 z8 = {0, 0, 0, 0, 0, 0, 0, 0};
    __builtin_amdgcn_tensor_load_to_lds(g0, g1, z, z, z8, 0);
#else
    __builtin_amdgcn_tensor_load_to_lds(g0, g1, z, z, 0);
#endif
}
#endif

#if !USE_TDM && USE_ASYNC
// Fallback path only compiled when TDM is unavailable.
__device__ __forceinline__ void async_cp16B(const half_t* g, half_t* l) {
    __builtin_amdgcn_global_load_async_to_lds_b128(
        (AS_GLOBAL v4i*)g, (AS_LDS v4i*)l, 0, 0);
}
__device__ __forceinline__ void wait_async_keep4() {
#if __has_builtin(__builtin_amdgcn_s_wait_asynccnt)
    __builtin_amdgcn_s_wait_asynccnt(4);
#else
    asm volatile("s_wait_asynccnt 0x4" ::: "memory");
#endif
}
__device__ __forceinline__ void wait_async_all() {
#if __has_builtin(__builtin_amdgcn_s_wait_asynccnt)
    __builtin_amdgcn_s_wait_asynccnt(0);
#else
    asm volatile("s_wait_asynccnt 0x0" ::: "memory");
#endif
}
#endif

// ---------------------------------------------------------------------------
// 1) Weights -> f16; W_hh rows permuted into gate-tiled order so the
//    recurrent A panel is a contiguous 128-row block:
//      WhhR[hb*128 + g*32 + j] = Whh[g*512 + hb*32 + j]
// ---------------------------------------------------------------------------
__global__ void k_convert_weights(const float* __restrict__ Wih,
                                  const float* __restrict__ Whh,
                                  const float* __restrict__ bih,
                                  const float* __restrict__ bhh,
                                  half_t* __restrict__ WihH,
                                  half_t* __restrict__ WhhR,
                                  float*  __restrict__ bias) {
    int i = blockIdx.x * blockDim.x + threadIdx.x;   // over 2048*512
    if (i < G4H * EE) {
        WihH[i] = (half_t)Wih[i];
        const int sr = i / EE;            // source row of W_hh
        const int k  = i % EE;
        const int g  = sr >> 9;           // gate
        const int hh = sr & 511;
        const int dr = (hh >> 5) * 128 + g * 32 + (hh & 31);
        WhhR[(size_t)dr * EE + k] = (half_t)Whh[i];
    }
    if (i < G4H) bias[i] = bih[i] + bhh[i];
}

// ---------------------------------------------------------------------------
// 2) Pack x row-major per timestep: Xt[t][n][e] = x[n][t][e]
// ---------------------------------------------------------------------------
__global__ void k_pack_x(const float* __restrict__ fp,
                         const float* __restrict__ ap,
                         half_t* __restrict__ Xt) {
    const size_t i = (size_t)blockIdx.x * blockDim.x + threadIdx.x; // t,n,e linear
    const int e = (int)(i % EE);
    const int n = (int)((i / EE) % NN);
    const int t = (int)(i / ((size_t)EE * NN));
    const int b = n >> 5;
    const int j = n & 31;
    const float* src = (j < 16)
        ? (fp + (((size_t)(b * SS + j)        * LL + t) * EE))
        : (ap + (((size_t)(b * SS + (j - 16)) * LL + t) * EE));
    Xt[i] = (half_t)src[e];
}

// ---------------------------------------------------------------------------
// 3) Zero h (f16 [n][h] + f32 [h][n]) and c
// ---------------------------------------------------------------------------
__global__ void k_init_state(half_t* __restrict__ hH,
                             float* __restrict__ hF,
                             float* __restrict__ cF) {
    int i = blockIdx.x * blockDim.x + threadIdx.x;
    if (i < HH * NN) { hH[i] = (half_t)0.0f; hF[i] = 0.0f; cF[i] = 0.0f; }
}

// ---------------------------------------------------------------------------
// Double-buffered WMMA mainloop: D(128 rows x 128 n) += A[rows][512] * B^T,
// A rows start at `arow0` in Aglob (row-major, ld=512); B rows are
// activations Bglob[brow0 + n][512] (row-major, ld=512).
// A/B tiles stream through LDS via TDM (preferred) / async-to-LDS / manual.
// ---------------------------------------------------------------------------
__device__ __forceinline__ void gemm_mainloop(
        const half_t* __restrict__ Aglob, int arow0, int arows_total,
        const half_t* __restrict__ Bglob, int brow0,
        half_t* __restrict__ sA0, half_t* __restrict__ sA1,
        half_t* __restrict__ sB0, half_t* __restrict__ sB1,
        v8f acc[4][2]) {
    const int tid  = threadIdx.x;
    const int lane = tid & 31;
    const int wave = tid >> 5;
    const int wr   = wave >> 2;          // 0..1 : 64-row block
    const int wc   = wave & 3;           // 0..3 : 32-col block
    const int l15  = lane & 15;
    const int lhi  = lane >> 4;

    const half_t* Abase = Aglob + (size_t)arow0 * EE;
    const half_t* Bbase = Bglob + (size_t)brow0 * EE;

#if USE_TDM
    if (wave == 0) {                      // issue chunk 0
        tdm_load_tile(Abase, lds_offset(sA0), arows_total);
        tdm_load_tile(Bbase, lds_offset(sB0), NN);
    }
#else
    const int srow = tid >> 1;                 // staged row (0..127)
    const int soff = (tid & 1) * 16;           // half offset in 32-k chunk
  #if USE_ASYNC
    {
        async_cp16B(Abase + (size_t)srow * EE + soff,     sA0 + srow * LPAD + soff);
        async_cp16B(Abase + (size_t)srow * EE + soff + 8, sA0 + srow * LPAD + soff + 8);
        async_cp16B(Bbase + (size_t)srow * EE + soff,     sB0 + srow * LPAD + soff);
        async_cp16B(Bbase + (size_t)srow * EE + soff + 8, sB0 + srow * LPAD + soff + 8);
    }
  #else
    {
        const half_t* a = Abase + (size_t)srow * EE + soff;
        const half_t* b = Bbase + (size_t)srow * EE + soff;
        #pragma unroll
        for (int j = 0; j < 16; ++j) { sA0[srow * LPAD + soff + j] = a[j];
                                       sB0[srow * LPAD + soff + j] = b[j]; }
    }
  #endif
#endif

    for (int ic = 0; ic < NCH; ++ic) {
        const int kk = ic * 32;
        half_t* cA = (ic & 1) ? sA1 : sA0;
        half_t* cB = (ic & 1) ? sB1 : sB0;
        half_t* nA = (ic & 1) ? sA0 : sA1;
        half_t* nB = (ic & 1) ? sB0 : sB1;

        // ---- issue next chunk's staging, then wait for current chunk ----
#if USE_TDM
        if (wave == 0) {
            if (ic + 1 < NCH) {
                tdm_load_tile(Abase + kk + 32, lds_offset(nA), arows_total);
                tdm_load_tile(Bbase + kk + 32, lds_offset(nB), NN);
                __builtin_amdgcn_s_wait_tensorcnt(2);
            } else {
                __builtin_amdgcn_s_wait_tensorcnt(0);
            }
        }
#elif USE_ASYNC
        if (ic + 1 < NCH) {
            const half_t* a = Abase + (size_t)srow * EE + kk + 32 + soff;
            const half_t* b = Bbase + (size_t)srow * EE + kk + 32 + soff;
            async_cp16B(a,     nA + srow * LPAD + soff);
            async_cp16B(a + 8, nA + srow * LPAD + soff + 8);
            async_cp16B(b,     nB + srow * LPAD + soff);
            async_cp16B(b + 8, nB + srow * LPAD + soff + 8);
            wait_async_keep4();
        } else {
            wait_async_all();
        }
#else
        if (ic + 1 < NCH) {
            const half_t* a = Abase + (size_t)srow * EE + kk + 32 + soff;
            const half_t* b = Bbase + (size_t)srow * EE + kk + 32 + soff;
            #pragma unroll
            for (int j = 0; j < 16; ++j) { nA[srow * LPAD + soff + j] = a[j];
                                           nB[srow * LPAD + soff + j] = b[j]; }
        }
#endif
        __syncthreads();

        // ---- fragments (ISA 16-bit A 16x32 / B 32x16 layouts) ----
        v16h af[4], bf[2];
        #pragma unroll
        for (int fa = 0; fa < 4; ++fa) {
            const half_t* base = cA + (wr * 64 + fa * 16 + l15) * LPAD;
            #pragma unroll
            for (int j = 0; j < 8; ++j) {
                af[fa][j]     = base[lhi * 8 + j];
                af[fa][8 + j] = base[16 + lhi * 8 + j];
            }
        }
        #pragma unroll
        for (int fb = 0; fb < 2; ++fb) {
            const half_t* base = cB + (wc * 32 + fb * 16 + l15) * LPAD;
            #pragma unroll
            for (int j = 0; j < 8; ++j) {
                bf[fb][j]     = base[lhi * 8 + j];
                bf[fb][8 + j] = base[16 + lhi * 8 + j];
            }
        }

        #pragma unroll
        for (int fa = 0; fa < 4; ++fa)
            #pragma unroll
            for (int fb = 0; fb < 2; ++fb)
                acc[fa][fb] = __builtin_amdgcn_wmma_f32_16x16x32_f16(
                    false, af[fa], false, bf[fb], (short)0, acc[fa][fb],
                    false, false);
        __syncthreads();
    }
}

// ---------------------------------------------------------------------------
// 4) Input GEMM: IG[t][col][n] = sum_e Xt[t][n][e]*Wih[col][e] + bias[col]
//    grid = (2048/128, 512/128, 64)
// ---------------------------------------------------------------------------
__global__ __launch_bounds__(256)
void k_input_gemm(const half_t* __restrict__ WihH,
                  const half_t* __restrict__ Xt,
                  const float* __restrict__ bias,
                  float* __restrict__ IG) {
    __shared__ half_t sA0[128 * LPAD], sA1[128 * LPAD];
    __shared__ half_t sB0[128 * LPAD], sB1[128 * LPAD];
    const int colbase = blockIdx.x * 128;
    const int n0      = blockIdx.y * 128;
    const int t       = blockIdx.z;

    v8f acc[4][2] = {};
    gemm_mainloop(WihH, colbase, G4H,
                  Xt + (size_t)t * NN * EE, n0,
                  sA0, sA1, sB0, sB1, acc);

    const int lane = threadIdx.x & 31;
    const int wave = threadIdx.x >> 5;
    const int wr = wave >> 2, wc = wave & 3;
    const int l15 = lane & 15, lhi = lane >> 4;
    #pragma unroll
    for (int fa = 0; fa < 4; ++fa)
        #pragma unroll
        for (int fb = 0; fb < 2; ++fb) {
            const int rowT = wr * 64 + fa * 16 + lhi * 8;
            const int nT   = n0 + wc * 32 + fb * 16 + l15;
            #pragma unroll
            for (int r = 0; r < 8; ++r) {
                const int col = colbase + rowT + r;
                IG[((size_t)t * G4H + col) * NN + nT] = acc[fa][fb][r] + bias[col];
            }
        }
}

// ---------------------------------------------------------------------------
// 5) One LSTM timestep: recurrent WMMA GEMM (A = gate-tiled WhhR) + cell
//    update.  Tile = 32 h x 128 n;  grid = (512/32, 512/128)
// ---------------------------------------------------------------------------
__global__ __launch_bounds__(256)
void k_lstm_step(const half_t* __restrict__ WhhR,
                 const half_t* __restrict__ hH_in,   // [n][h] f16
                 const float* __restrict__ IG,
                 float* __restrict__ cF,             // [h][n]
                 float* __restrict__ hF,             // [h][n]
                 half_t* __restrict__ hH_out,        // [n][h]
                 int t) {
    __shared__ half_t sA0[128 * LPAD], sA1[128 * LPAD];
    __shared__ half_t sB0[128 * LPAD], sB1[128 * LPAD];
    __shared__ float  sG[128 * GPAD];
    const int h0 = blockIdx.x * 32;
    const int n0 = blockIdx.y * 128;

    v8f acc[4][2] = {};
    gemm_mainloop(WhhR, blockIdx.x * 128, G4H,
                  hH_in, n0,
                  sA0, sA1, sB0, sB1, acc);

    const int tid  = threadIdx.x;
    const int lane = tid & 31;
    const int wave = tid >> 5;
    const int wr = wave >> 2, wc = wave & 3;
    const int l15 = lane & 15, lhi = lane >> 4;

    // gate pre-activations + precomputed input gates -> LDS
    #pragma unroll
    for (int fa = 0; fa < 4; ++fa)
        #pragma unroll
        for (int fb = 0; fb < 2; ++fb) {
            const int rowT = wr * 64 + fa * 16 + lhi * 8;
            const int nl   = wc * 32 + fb * 16 + l15;
            #pragma unroll
            for (int r = 0; r < 8; ++r) {
                const int row = rowT + r;             // gate = row>>5, h = row&31
                const int col = (row >> 5) * HH + h0 + (row & 31);
                const float ig = IG[((size_t)t * G4H + col) * NN + n0 + nl];
                sG[row * GPAD + nl] = acc[fa][fb][r] + ig;
            }
        }
    __syncthreads();

    // cell update: 32h x 128n = 4096 elements, 16 per thread
    #pragma unroll
    for (int e = 0; e < 16; ++e) {
        const int p  = tid * 16 + e;
        const int hh = p >> 7;
        const int nn = p & 127;
        const float gi = sigf (sG[( 0 + hh) * GPAD + nn]);
        const float gf = sigf (sG[(32 + hh) * GPAD + nn]);
        const float gg = tanhf(sG[(64 + hh) * GPAD + nn]);
        const float go = sigf (sG[(96 + hh) * GPAD + nn]);
        const size_t gidx = (size_t)(h0 + hh) * NN + n0 + nn;   // [h][n]
        const float c = gf * cF[gidx] + gi * gg;
        cF[gidx] = c;
        const float h = go * tanhf(c);
        hF[gidx] = h;
        hH_out[(size_t)(n0 + nn) * HH + h0 + hh] = (half_t)h;   // [n][h]
    }
}

// ---------------------------------------------------------------------------
// 6) Pool over S (mean + max, both sides) -> feat[b][4H]; hF is [h][n]
// ---------------------------------------------------------------------------
__global__ void k_pool(const float* __restrict__ hF, float* __restrict__ feat) {
    int i = blockIdx.x * blockDim.x + threadIdx.x;   // 16*512
    if (i >= BB * HH) return;
    const int b = i / HH;
    const int h = i % HH;
    const float* row = hF + (size_t)h * NN + b * 32;
    float sm0 = 0.f, mx0 = -3.4e38f, sm1 = 0.f, mx1 = -3.4e38f;
    #pragma unroll
    for (int s = 0; s < SS; ++s) {
        float v0 = row[s];
        float v1 = row[16 + s];
        sm0 += v0; mx0 = fmaxf(mx0, v0);
        sm1 += v1; mx1 = fmaxf(mx1, v1);
    }
    float* fb = feat + (size_t)b * G4H;
    fb[h]          = sm0 * (1.0f / SS);
    fb[HH + h]     = mx0;
    fb[2 * HH + h] = sm1 * (1.0f / SS);
    fb[3 * HH + h] = mx1;
}

// ---------------------------------------------------------------------------
// 7) fc1 -> fc2 -> sigmoid (no activation between, per reference)
// ---------------------------------------------------------------------------
__global__ void k_fc(const float* __restrict__ feat,
                     const float* __restrict__ fc1w, const float* __restrict__ fc1b,
                     const float* __restrict__ fc2w, const float* __restrict__ fc2b,
                     float* __restrict__ out) {
    __shared__ float sZ[BB * O1];
    const int j = threadIdx.x;               // 0..255
    const float* wv = fc1w + (size_t)j * G4H;
    for (int b = 0; b < BB; ++b) {
        const float* fv = feat + (size_t)b * G4H;
        float a = 0.f;
        for (int k = 0; k < G4H; ++k) a += fv[k] * wv[k];
        sZ[b * O1 + j] = a + fc1b[j];
    }
    __syncthreads();
    if (j < BB) {
        float a = fc2b[0];
        for (int k = 0; k < O1; ++k) a += sZ[j * O1 + k] * fc2w[k];
        out[j] = sigf(a);
    }
}

// ---------------------------------------------------------------------------
// Launch
// ---------------------------------------------------------------------------
extern "C" void kernel_launch(void* const* d_in, const int* in_sizes, int n_in,
                              void* d_out, int out_size, void* d_ws, size_t ws_size,
                              hipStream_t stream) {
    const float* fp   = (const float*)d_in[0];
    const float* ap   = (const float*)d_in[1];
    const float* Wih  = (const float*)d_in[2];
    const float* Whh  = (const float*)d_in[3];
    const float* bih  = (const float*)d_in[4];
    const float* bhh  = (const float*)d_in[5];
    const float* f1w  = (const float*)d_in[6];
    const float* f1b  = (const float*)d_in[7];
    const float* f2w  = (const float*)d_in[8];
    const float* f2b  = (const float*)d_in[9];
    float* out = (float*)d_out;

    char* p = (char*)d_ws;
    half_t* WihH = (half_t*)p;            p += (size_t)G4H * EE * 2;        // 2 MB
    half_t* WhhR = (half_t*)p;            p += (size_t)G4H * EE * 2;        // 2 MB
    float*  bias = (float*)p;             p += (size_t)G4H * 4;             // 8 KB
    half_t* Xt   = (half_t*)p;            p += (size_t)LL * NN * EE * 2;    // 33.5 MB
    float*  IG   = (float*)p;             p += (size_t)LL * G4H * NN * 4;   // 268 MB
    half_t* hH   = (half_t*)p;            p += (size_t)NN * HH * 2;         // 0.5 MB
    float*  hF   = (float*)p;             p += (size_t)HH * NN * 4;         // 1 MB
    float*  cF   = (float*)p;             p += (size_t)HH * NN * 4;         // 1 MB
    float*  feat = (float*)p;             p += (size_t)BB * G4H * 4;        // 128 KB

    k_convert_weights<<<(G4H * EE + 255) / 256, 256, 0, stream>>>(
        Wih, Whh, bih, bhh, WihH, WhhR, bias);

    k_pack_x<<<(int)(((size_t)LL * NN * EE + 255) / 256), 256, 0, stream>>>(fp, ap, Xt);

    k_init_state<<<(HH * NN + 255) / 256, 256, 0, stream>>>(hH, hF, cF);

    k_input_gemm<<<dim3(G4H / 128, NN / 128, LL), 256, 0, stream>>>(WihH, Xt, bias, IG);

    for (int t = 0; t < LL; ++t)
        k_lstm_step<<<dim3(HH / 32, NN / 128), 256, 0, stream>>>(
            WhhR, hH, IG, cF, hF, hH, t);

    k_pool<<<(BB * HH + 255) / 256, 256, 0, stream>>>(hF, feat);
    k_fc<<<1, 256, 0, stream>>>(feat, f1w, f1b, f2w, f2b, out);
}
